// GIN_DISMAT_19473381720872
// MI455X (gfx1250) — matrix operand order repaired and checked by reading, not compiled, not run on hardware
//
#include <hip/hip_runtime.h>
#include <hip/hip_bf16.h>

// ---------------------------------------------------------------------------
// GIN forward for MI455X (gfx1250): bf16 WMMA GEMMs (packed-B fragments) +
// f32 atomic edge scatter.
// ---------------------------------------------------------------------------

typedef __attribute__((ext_vector_type(16))) __bf16 v16bf;
typedef __attribute__((ext_vector_type(8)))  __bf16 v8bf;
typedef __attribute__((ext_vector_type(8)))  float  v8f;

static __device__ __forceinline__ __bf16 f2bf(float f) {
    unsigned u = __float_as_uint(f);
    unsigned r = (u + 0x7FFFu + ((u >> 16) & 1u)) >> 16;   // round-to-nearest-even
    unsigned short s = (unsigned short)r;
    return __builtin_bit_cast(__bf16, s);
}

static __device__ __forceinline__ float elu(float v) {
    return v > 0.0f ? v : (__expf(v) - 1.0f);
}

// -------------------- elementwise helpers ----------------------------------

__global__ void k_copy_f4(const float* __restrict__ in, float* __restrict__ out,
                          long long n4) {
    long long t = (long long)blockIdx.x * blockDim.x + threadIdx.x;
    if (t < n4) ((float4*)out)[t] = ((const float4*)in)[t];
}

__global__ void k_zero_f32(float* __restrict__ p, int n) {
    int t = blockIdx.x * blockDim.x + threadIdx.x;
    if (t < n) p[t] = 0.0f;
}

__global__ void k_f32_to_bf16(const float* __restrict__ in, __bf16* __restrict__ out,
                              long long n) {
    long long t = (long long)blockIdx.x * blockDim.x + threadIdx.x;
    if (t < n) out[t] = f2bf(in[t]);
}

// Pack a row-major f32 weight B[K x N] into bf16 WMMA-B fragment order:
//   out[(((ntile * (K/32) + kk32) * 32) + lane) * 16 + e]
//     = B[(kk32*32 + (e<8 ? (lane/16)*8 + e : 16 + (lane/16)*8 + e-8)) * N + ntile*16 + (lane%16)]
// so each lane's 16-value fragment is one contiguous 32-byte load.
__global__ void k_pack_b(const float* __restrict__ B, __bf16* __restrict__ out,
                         int K, int N) {
    long long t = (long long)blockIdx.x * blockDim.x + threadIdx.x;
    long long total = (long long)K * N;
    if (t >= total) return;
    const int K32 = K >> 5;
    int e     = (int)(t & 15);
    int lane  = (int)((t >> 4) & 31);
    long long grp = t >> 9;                 // ntile * K32 + kk32
    int kk32  = (int)(grp % K32);
    int ntile = (int)(grp / K32);
    int half  = lane >> 4;
    int r     = lane & 15;
    int k     = kk32 * 32 + (e < 8 ? half * 8 + e : 16 + half * 8 + (e - 8));
    int col   = ntile * 16 + r;
    out[t] = f2bf(B[(size_t)k * N + col]);
}

// -------------------- edge scatter-add (agg += x[src] at dst) --------------

__global__ void k_scatter_edges(const float* __restrict__ x,
                                const int* __restrict__ src,
                                const int* __restrict__ dst,
                                float* __restrict__ agg,
                                int dim4, long long total) {
    long long t = (long long)blockIdx.x * blockDim.x + threadIdx.x;
    if (t >= total) return;
    int e = (int)(t / dim4);
    int c = (int)(t % dim4);
    int dim = dim4 * 4;
    float4 v = ((const float4*)(x + (size_t)src[e] * dim))[c];
    float* dp = agg + (size_t)dst[e] * dim + (size_t)c * 4;
    __hip_atomic_fetch_add(dp + 0, v.x, __ATOMIC_RELAXED, __HIP_MEMORY_SCOPE_AGENT);
    __hip_atomic_fetch_add(dp + 1, v.y, __ATOMIC_RELAXED, __HIP_MEMORY_SCOPE_AGENT);
    __hip_atomic_fetch_add(dp + 2, v.z, __ATOMIC_RELAXED, __HIP_MEMORY_SCOPE_AGENT);
    __hip_atomic_fetch_add(dp + 3, v.w, __ATOMIC_RELAXED, __HIP_MEMORY_SCOPE_AGENT);
}

// -------------------- graph pooling ----------------------------------------

__global__ void k_pool(const float* __restrict__ x, const int* __restrict__ batch,
                       float* __restrict__ pooled, int n_nodes) {
    long long t = (long long)blockIdx.x * blockDim.x + threadIdx.x;
    long long total = (long long)n_nodes * 32;     // 128 feats / 4
    if (t >= total) return;
    int node = (int)(t >> 5);
    int c    = (int)(t & 31);
    float4 v = ((const float4*)(x + (size_t)node * 128))[c];
    float* dp = pooled + (size_t)batch[node] * 128 + (size_t)c * 4;
    __hip_atomic_fetch_add(dp + 0, v.x, __ATOMIC_RELAXED, __HIP_MEMORY_SCOPE_AGENT);
    __hip_atomic_fetch_add(dp + 1, v.y, __ATOMIC_RELAXED, __HIP_MEMORY_SCOPE_AGENT);
    __hip_atomic_fetch_add(dp + 2, v.z, __ATOMIC_RELAXED, __HIP_MEMORY_SCOPE_AGENT);
    __hip_atomic_fetch_add(dp + 3, v.w, __ATOMIC_RELAXED, __HIP_MEMORY_SCOPE_AGENT);
}

// -------------------- WMMA bf16 GEMM ----------------------------------------
// C[M,N] = epilogue(A[M,K] @ B[K,N] + bias), B pre-packed by k_pack_b.
// Each wave computes a 16x64 strip: one A fragment reused by 4 B fragments /
// 4 accumulators per K-step. 8 waves stack in M -> 128x64 block tile.
// MODE 0: ELU, store bf16 ; MODE 1: BN(ELU(.)), store f32 ; MODE 2: id, f32.

template <int MODE>
__global__ __launch_bounds__(256)
void k_gemm_bf16(const __bf16* __restrict__ A, const __bf16* __restrict__ Bpk,
                 const float* __restrict__ bias, int M, int N, int K,
                 float* __restrict__ outf, __bf16* __restrict__ outb,
                 const float* __restrict__ gamma, const float* __restrict__ beta,
                 const float* __restrict__ mean, const float* __restrict__ var) {
    const int wave = threadIdx.x >> 5;
    const int lane = threadIdx.x & 31;
    const int half = lane >> 4;          // lane group (0: lanes 0-15, 1: 16-31)
    const int r    = lane & 15;

    const int tileM = blockIdx.y * 128 + wave * 16;
    if (tileM >= M) return;
    const int ntile0 = blockIdx.x * 4;   // four 16-wide column tiles
    const int K32 = K >> 5;

    const int arow = tileM + r;          // A: lane%16 = row
    const int akb  = half * 8;           // A K-base per ISA 16-bit A layout

    v8f acc0 = {}, acc1 = {}, acc2 = {}, acc3 = {};
    for (int kk32 = 0; kk32 < K32; ++kk32) {
        const __bf16* Ap = A + (size_t)arow * K + kk32 * 32 + akb;
        v8bf alo = *(const v8bf*)(Ap);        // K = akb + 0..7
        v8bf ahi = *(const v8bf*)(Ap + 16);   // K = akb + 16..23
        v16bf a;
#pragma unroll
        for (int j = 0; j < 8; ++j) { a[j] = alo[j]; a[8 + j] = ahi[j]; }

        const __bf16* Bp =
            Bpk + ((((size_t)ntile0 * K32 + kk32) * 32 + lane) << 4);
        const size_t tstride = (size_t)K32 * 512;   // one ntile of packed B
        v16bf b0 = *(const v16bf*)(Bp);
        v16bf b1 = *(const v16bf*)(Bp + tstride);
        v16bf b2 = *(const v16bf*)(Bp + 2 * tstride);
        v16bf b3 = *(const v16bf*)(Bp + 3 * tstride);

        acc0 = __builtin_amdgcn_wmma_f32_16x16x32_bf16(false, a, false, b0,
                                                       (short)0, acc0, false, false);
        acc1 = __builtin_amdgcn_wmma_f32_16x16x32_bf16(false, a, false, b1,
                                                       (short)0, acc1, false, false);
        acc2 = __builtin_amdgcn_wmma_f32_16x16x32_bf16(false, a, false, b2,
                                                       (short)0, acc2, false, false);
        acc3 = __builtin_amdgcn_wmma_f32_16x16x32_bf16(false, a, false, b3,
                                                       (short)0, acc3, false, false);
    }

    v8f accs[4] = {acc0, acc1, acc2, acc3};
#pragma unroll
    for (int t = 0; t < 4; ++t) {
        const int col = (ntile0 + t) * 16 + r;
        const float bi = bias[col];
        float sc = 1.0f, sh = 0.0f;
        if (MODE == 1) {
            sc = gamma[col] * rsqrtf(var[col] + 1e-5f);
            sh = beta[col] - mean[col] * sc;
        }
#pragma unroll
        for (int v = 0; v < 8; ++v) {
            const int row = tileM + half * 8 + v;   // C/D layout: M = 8*half + vgpr
            float val = accs[t][v] + bi;
            if (MODE == 0 || MODE == 1) val = elu(val);
            if (MODE == 1) val = val * sc + sh;
            if (MODE == 0) outb[(size_t)row * N + col] = f2bf(val);
            else           outf[(size_t)row * N + col] = val;
        }
    }
}

// -------------------- final outer-product expansion -------------------------
// out[b,i,j,f] = pooled[b,f] * g[b,i] * g[b,j]   (B=128, DG=64, F=128)

__global__ void k_outer(const float* __restrict__ pooled, const float* __restrict__ g,
                        float* __restrict__ out) {
    long long t = (long long)blockIdx.x * blockDim.x + threadIdx.x;  // one float4
    int f4 = (int)(t & 31);
    int j  = (int)((t >> 5) & 63);
    int i  = (int)((t >> 11) & 63);
    int b  = (int)(t >> 17);
    float s = g[b * 64 + i] * g[b * 64 + j];
    float4 p = ((const float4*)(pooled + (size_t)b * 128))[f4];
    float4 o = make_float4(p.x * s, p.y * s, p.z * s, p.w * s);
    ((float4*)out)[t] = o;
}

// ---------------------------------------------------------------------------

extern "C" void kernel_launch(void* const* d_in, const int* in_sizes, int n_in,
                              void* d_out, int out_size, void* d_ws, size_t ws_size,
                              hipStream_t stream) {
    const float* x     = (const float*)d_in[0];
    const int*   ei    = (const int*)d_in[1];
    const int*   batch = (const int*)d_in[2];

    const int NN = in_sizes[2];          // 100000 nodes
    const int NE = in_sizes[1] / 2;      // 3200000 edges
    const int NG = 128;                  // graphs
    const int* src = ei;
    const int* dst = ei + NE;

    const float *W1[5], *B1[5], *W2[5], *B2[5], *GM[5], *BT[5], *MN[5], *VR[5];
    for (int l = 0; l < 5; ++l) {
        const int base = 3 + 8 * l;
        W1[l] = (const float*)d_in[base + 0];
        B1[l] = (const float*)d_in[base + 1];
        W2[l] = (const float*)d_in[base + 2];
        B2[l] = (const float*)d_in[base + 3];
        GM[l] = (const float*)d_in[base + 4];
        BT[l] = (const float*)d_in[base + 5];
        MN[l] = (const float*)d_in[base + 6];
        VR[l] = (const float*)d_in[base + 7];
    }
    const float* fc1W = (const float*)d_in[43];
    const float* fc1b = (const float*)d_in[44];
    const float* fc2W = (const float*)d_in[45];
    const float* fc2b = (const float*)d_in[46];

    const int din[5]  = {128, 256, 256, 256, 256};
    const int dout[5] = {256, 256, 256, 256, 128};
    const int DH = 256;

    // ---- workspace carve-out ----
    char* ws = (char*)d_ws;
    size_t off = 0;
    auto alloc = [&](size_t bytes) -> void* {
        void* p = ws + off;
        off += (bytes + 255) & ~(size_t)255;
        return p;
    };
    float*  agg      = (float*)alloc((size_t)NN * 256 * 4);
    __bf16* Abf      = (__bf16*)alloc((size_t)NN * 256 * 2);
    __bf16* Tbf      = (__bf16*)alloc((size_t)NN * 256 * 2);
    float*  Xf       = (float*)alloc((size_t)NN * 256 * 4);
    __bf16* W1bf[5], *W2bf[5];
    for (int l = 0; l < 5; ++l) {
        W1bf[l] = (__bf16*)alloc((size_t)din[l] * DH * 2);
        W2bf[l] = (__bf16*)alloc((size_t)DH * dout[l] * 2);
    }
    __bf16* fc1Wbf   = (__bf16*)alloc(128 * 128 * 2);
    __bf16* fc2Wbf   = (__bf16*)alloc(128 * 64 * 2);
    float*  pooled   = (float*)alloc(NG * 128 * 4);
    __bf16* pooledbf = (__bf16*)alloc(NG * 128 * 2);
    __bf16* hbf      = (__bf16*)alloc(NG * 128 * 2);
    float*  gbuf     = (float*)alloc(NG * 64 * 4);

    auto cvt = [&](const float* in, __bf16* out, long long n) {
        unsigned blocks = (unsigned)((n + 255) / 256);
        k_f32_to_bf16<<<blocks, 256, 0, stream>>>(in, out, n);
    };
    auto packB = [&](const float* B, __bf16* out, int K, int N) {
        long long n = (long long)K * N;
        unsigned blocks = (unsigned)((n + 255) / 256);
        k_pack_b<<<blocks, 256, 0, stream>>>(B, out, K, N);
    };

    // ---- weight packing (cheap, per launch, deterministic) ----
    for (int l = 0; l < 5; ++l) {
        packB(W1[l], W1bf[l], din[l], DH);
        packB(W2[l], W2bf[l], DH, dout[l]);
    }
    packB(fc1W, fc1Wbf, 128, 128);
    packB(fc2W, fc2Wbf, 128, 64);

    // ---- GIN layers ----
    const float* xin = x;
    for (int l = 0; l < 5; ++l) {
        const int d = din[l];
        // agg = x  (so agg after scatter holds x + sum_{j->i} x_j)
        {
            long long n4 = (long long)NN * d / 4;
            unsigned blocks = (unsigned)((n4 + 255) / 256);
            k_copy_f4<<<blocks, 256, 0, stream>>>(xin, agg, n4);
        }
        // agg[dst] += x[src]
        {
            long long total = (long long)NE * (d / 4);
            unsigned blocks = (unsigned)((total + 255) / 256);
            k_scatter_edges<<<blocks, 256, 0, stream>>>(xin, src, dst, agg, d / 4, total);
        }
        // bf16 copy of (x + agg)
        cvt(agg, Abf, (long long)NN * d);
        // T = ELU(A @ W1 + b1)  -> bf16
        {
            dim3 grid(DH / 64, (NN + 127) / 128);
            k_gemm_bf16<0><<<grid, 256, 0, stream>>>(Abf, W1bf[l], B1[l], NN, DH, d,
                                                     nullptr, Tbf,
                                                     nullptr, nullptr, nullptr, nullptr);
        }
        // X = BN(ELU(T @ W2 + b2)) -> f32
        {
            dim3 grid(dout[l] / 64, (NN + 127) / 128);
            k_gemm_bf16<1><<<grid, 256, 0, stream>>>(Tbf, W2bf[l], B2[l], NN, dout[l], DH,
                                                     Xf, nullptr,
                                                     GM[l], BT[l], MN[l], VR[l]);
        }
        xin = Xf;
    }

    // ---- pooling ----
    k_zero_f32<<<(NG * 128 + 255) / 256, 256, 0, stream>>>(pooled, NG * 128);
    {
        long long total = (long long)NN * 32;
        unsigned blocks = (unsigned)((total + 255) / 256);
        k_pool<<<blocks, 256, 0, stream>>>(Xf, batch, pooled, NN);
    }
    cvt(pooled, pooledbf, NG * 128);

    // ---- fc head ----
    {
        dim3 grid(128 / 64, 1);
        k_gemm_bf16<0><<<grid, 256, 0, stream>>>(pooledbf, fc1Wbf, fc1b, NG, 128, 128,
                                                 nullptr, hbf,
                                                 nullptr, nullptr, nullptr, nullptr);
    }
    {
        dim3 grid(64 / 64, 1);
        k_gemm_bf16<2><<<grid, 256, 0, stream>>>(hbf, fc2Wbf, fc2b, NG, 64, 128,
                                                 gbuf, nullptr,
                                                 nullptr, nullptr, nullptr, nullptr);
    }

    // ---- out[b,i,j,f] = pooled[b,f] * g[b,i] * g[b,j] ----
    {
        long long total4 = (long long)NG * 64 * 64 * 32;   // 16777216 float4
        unsigned blocks = (unsigned)(total4 / 256);
        k_outer<<<blocks, 256, 0, stream>>>(pooled, gbuf, (float*)d_out);
    }
}
